// NemotronHExperts_6605659701708
// MI455X (gfx1250) — matrix operand order repaired and checked by reading, not compiled, not run on hardware
//
#include <hip/hip_runtime.h>
#include <hip/hip_bf16.h>

// ---------------- problem constants (match reference) ----------------
#define TOKENS 8192
#define HIDDEN 2048
#define INTER  4096
#define NEXP   8

// ---------------- tiling ----------------
#define MT 128          // token tile
#define NT 128          // output-col tile
#define KT 32           // k step (one bf16 WMMA K)
#define LDS_STRIDE 40   // bf16 elements per LDS row (80 bytes, 16B-aligned rows)

typedef __attribute__((ext_vector_type(16))) __bf16 v16bf;
typedef __attribute__((ext_vector_type(8)))  __bf16 v8bf;
typedef __attribute__((ext_vector_type(8)))  float  v8f;

union V16u { v16bf v; v8bf h[2]; };

// A fragment: 16x32 bf16 (M x K). lane L: m = L&15; lanes 0-15 hold K 0..7,16..23;
// lanes 16-31 hold K 8..15,24..31  (two 16-byte LDS loads).
__device__ inline v16bf load_a_frag(const __bf16* lds, int mrow, int lane) {
  const __bf16* row = lds + (size_t)(mrow + (lane & 15)) * LDS_STRIDE;
  const int base = (lane >> 4) << 3;   // 0 or 8
  V16u r;
  r.h[0] = *(const v8bf*)(row + base);
  r.h[1] = *(const v8bf*)(row + base + 16);
  return r.v;
}

// B fragment: 32x16 bf16 (K x N). lane L: n = L&15; lanes 0-15 hold K 0..7,16..23;
// lanes 16-31 hold K 8..15,24..31 (two 16-byte LDS loads).
__device__ inline v16bf load_b_frag(const __bf16* lds, int nrow, int lane) {
  const __bf16* row = lds + (size_t)(nrow + (lane & 15)) * LDS_STRIDE;
  const int kbase = (lane >> 4) << 3;  // 0 or 8
  V16u r;
  r.h[0] = *(const v8bf*)(row + kbase);
  r.h[1] = *(const v8bf*)(row + kbase + 16);
  return r.v;
}

__device__ inline void cvt_store4(__bf16* dst, float4 v) {
  dst[0] = (__bf16)v.x; dst[1] = (__bf16)v.y;
  dst[2] = (__bf16)v.z; dst[3] = (__bf16)v.w;
}

// gfx1250 async copy: 16 bytes global -> LDS per lane, tracked by ASYNCcnt.
__device__ inline void async_copy_b128(unsigned lds_off, const void* gsrc) {
  asm volatile("global_load_async_to_lds_b128 %0, %1, off"
               :: "v"(lds_off), "v"(gsrc)
               : "memory");
}
__device__ inline void wait_async0() {
  asm volatile("s_wait_asynccnt 0x0" ::: "memory");
}
__device__ inline unsigned lds_off_of(const void* p) {
  // low 32 bits of a generic shared pointer == LDS byte offset
  return (unsigned)(unsigned long long)p;
}

// ---------------- kernel 1: zero output + routing counters ----------------
__global__ void moe_zero_kernel(float* __restrict__ out, int n, int* __restrict__ cnt) {
  int i = blockIdx.x * blockDim.x + threadIdx.x;
  if (i < n) out[i] = 0.0f;
  if (i < NEXP) cnt[i] = 0;
}

// ---------------- kernel 2: convert X (f32) -> Xb (bf16) ----------------
__global__ void moe_cvtx_kernel(const float* __restrict__ X, __bf16* __restrict__ Xb, int n4) {
  int i = blockIdx.x * blockDim.x + threadIdx.x;
  if (i < n4) {
    float4 v = *(const float4*)(X + (size_t)i * 4);
    cvt_store4(Xb + (size_t)i * 4, v);
  }
}

// ---------------- kernel 3: routing (bucket tokens per expert) ----------------
__global__ void moe_route_kernel(const int* __restrict__ idx, const float* __restrict__ wts,
                                 int* __restrict__ cnt, int* __restrict__ rowToken,
                                 float* __restrict__ rowW) {
  int t = blockIdx.x * blockDim.x + threadIdx.x;
  if (t >= TOKENS) return;
  int   i0 = idx[2 * t], i1 = idx[2 * t + 1];
  float w0 = wts[2 * t], w1 = wts[2 * t + 1];
  if (i0 == i1) {
    int p = atomicAdd(&cnt[i0], 1);
    rowToken[i0 * TOKENS + p] = t;
    rowW[i0 * TOKENS + p] = w0 + w1;
  } else {
    int p0 = atomicAdd(&cnt[i0], 1);
    rowToken[i0 * TOKENS + p0] = t;
    rowW[i0 * TOKENS + p0] = w0;
    int p1 = atomicAdd(&cnt[i1], 1);
    rowToken[i1 * TOKENS + p1] = t;
    rowW[i1 * TOKENS + p1] = w1;
  }
}

// ---------------- kernel 4: tiny exclusive prefix over 8 counts ----------------
__global__ void moe_prefix_kernel(const int* __restrict__ cnt, int* __restrict__ rowStart) {
  if (threadIdx.x == 0 && blockIdx.x == 0) {
    int s = 0;
    for (int e = 0; e < NEXP; ++e) { rowStart[e] = s; s += cnt[e]; }
  }
}

// ---------------- kernel 5: up-proj GEMM + relu^2, H stored bf16 ----------------
// grid: (expert, INTER/NT, TOKENS/MT), block 256 = 8 wave32 waves
__global__ __launch_bounds__(256) void moe_up_kernel(
    const __bf16* __restrict__ Xb, const float* __restrict__ Wup,
    const int* __restrict__ cnt, const int* __restrict__ rowStart,
    const int* __restrict__ rowToken, __bf16* __restrict__ Hbuf) {
  const int e     = blockIdx.x;
  const int ntile = blockIdx.y;
  const int mtile = blockIdx.z;
  const int rows  = cnt[e];
  const int mbase = mtile * MT;
  if (mbase >= rows) return;
  const int ncol0  = ntile * NT;
  const int rstart = rowStart[e];
  const int* tok   = rowToken + e * TOKENS;
  const float* W   = Wup + (size_t)e * INTER * HIDDEN;

  __shared__ alignas(32) __bf16 lA[2][MT * LDS_STRIDE];
  __shared__ alignas(32) __bf16 lB[2][NT * LDS_STRIDE];

  const int tid  = threadIdx.x;
  const int lane = tid & 31;
  const int wave = tid >> 5;
  const int wm   = wave & 1;          // 64-row half
  const int wn   = wave >> 1;         // 32-col quarter
  const int rowq = tid >> 3;          // 0..31 (B f32 staging)
  const int colq = (tid & 7) * 4;
  const int arow = tid >> 2;          // 0..63 (A bf16 async staging)
  const int achk = (tid & 3) * 8;     // bf16 chunk of 8 (16 bytes)

  // token per A-staging row, clamped (garbage rows filtered in epilogue)
  int tokA[2];
#pragma unroll
  for (int rr = 0; rr < 2; ++rr) {
    int gr = mbase + rr * 64 + arow;
    tokA[rr] = tok[gr < rows ? gr : (rows - 1)];
  }

  v8f acc[4][2] = {};
  float4 bReg[4];

  // ---- prologue: stage k0 = 0 into buffer 0 ----
#pragma unroll
  for (int rr = 0; rr < 2; ++rr)
    async_copy_b128(lds_off_of(&lA[0][(rr * 64 + arow) * LDS_STRIDE + achk]),
                    Xb + (size_t)tokA[rr] * HIDDEN + achk);
#pragma unroll
  for (int rr = 0; rr < 4; ++rr)
    bReg[rr] = *(const float4*)(W + (size_t)(ncol0 + rr * 32 + rowq) * HIDDEN + colq);
#pragma unroll
  for (int rr = 0; rr < 4; ++rr)
    cvt_store4(&lB[0][(rr * 32 + rowq) * LDS_STRIDE + colq], bReg[rr]);
  wait_async0();
  __syncthreads();

  int p = 0;
  for (int k0 = 0; k0 < HIDDEN; k0 += KT) {
    const int kn = k0 + KT;
    const bool more = (kn < HIDDEN);
    if (more) {
      // issue next-tile fills while this tile computes
#pragma unroll
      for (int rr = 0; rr < 2; ++rr)
        async_copy_b128(lds_off_of(&lA[p ^ 1][(rr * 64 + arow) * LDS_STRIDE + achk]),
                        Xb + (size_t)tokA[rr] * HIDDEN + kn + achk);
#pragma unroll
      for (int rr = 0; rr < 4; ++rr) {
        const float* wp = W + (size_t)(ncol0 + rr * 32 + rowq) * HIDDEN + kn + colq;
        bReg[rr] = *(const float4*)wp;
        __builtin_prefetch(wp + KT, 0, 0);
      }
    }
    // compute current tile
    v16bf af[4], bfr[2];
#pragma unroll
    for (int mf = 0; mf < 4; ++mf) af[mf] = load_a_frag(lA[p], wm * 64 + mf * 16, lane);
#pragma unroll
    for (int nf = 0; nf < 2; ++nf) bfr[nf] = load_b_frag(lB[p], wn * 32 + nf * 16, lane);
#pragma unroll
    for (int mf = 0; mf < 4; ++mf)
#pragma unroll
      for (int nf = 0; nf < 2; ++nf)
        acc[mf][nf] = __builtin_amdgcn_wmma_f32_16x16x32_bf16(
            false, af[mf], false, bfr[nf], (short)0, acc[mf][nf], false, false);
    if (more) {
#pragma unroll
      for (int rr = 0; rr < 4; ++rr)
        cvt_store4(&lB[p ^ 1][(rr * 32 + rowq) * LDS_STRIDE + colq], bReg[rr]);
    }
    wait_async0();
    __syncthreads();
    p ^= 1;
  }

  // relu^2 then store H as bf16 (C/D layout: n = L&15, vgpr r -> m = r + 8*(L>>4))
#pragma unroll
  for (int mf = 0; mf < 4; ++mf) {
#pragma unroll
    for (int nf = 0; nf < 2; ++nf) {
      int n = ncol0 + wn * 32 + nf * 16 + (lane & 15);
#pragma unroll
      for (int r = 0; r < 8; ++r) {
        int m = wm * 64 + mf * 16 + r + ((lane >> 4) << 3);
        if (mbase + m < rows) {
          float v = fmaxf(acc[mf][nf][r], 0.0f);
          v = v * v;
          Hbuf[(size_t)(rstart + mbase + m) * INTER + n] = (__bf16)v;
        }
      }
    }
  }
}

// ---------------- kernel 6: down-proj GEMM, scale, scatter-add ----------------
// grid: (expert, HIDDEN/NT, TOKENS/MT), block 256
__global__ __launch_bounds__(256) void moe_down_kernel(
    const __bf16* __restrict__ Hbuf, const float* __restrict__ Wdown,
    const int* __restrict__ cnt, const int* __restrict__ rowStart,
    const int* __restrict__ rowToken, const float* __restrict__ rowW,
    float* __restrict__ out) {
  const int e     = blockIdx.x;
  const int ntile = blockIdx.y;
  const int mtile = blockIdx.z;
  const int rows  = cnt[e];
  const int mbase = mtile * MT;
  if (mbase >= rows) return;
  const int ncol0  = ntile * NT;
  const int rstart = rowStart[e];
  const float* W   = Wdown + (size_t)e * HIDDEN * INTER;

  __shared__ alignas(32) __bf16 lA[2][MT * LDS_STRIDE];
  __shared__ alignas(32) __bf16 lB[2][NT * LDS_STRIDE];

  const int tid  = threadIdx.x;
  const int lane = tid & 31;
  const int wave = tid >> 5;
  const int wm   = wave & 1;
  const int wn   = wave >> 1;
  const int rowq = tid >> 3;
  const int colq = (tid & 7) * 4;
  const int arow = tid >> 2;
  const int achk = (tid & 3) * 8;

  // clamped H row per A-staging row
  const __bf16* aSrc[2];
#pragma unroll
  for (int rr = 0; rr < 2; ++rr) {
    int gr = mbase + rr * 64 + arow;
    gr = gr < rows ? gr : (rows - 1);
    aSrc[rr] = Hbuf + (size_t)(rstart + gr) * INTER + achk;
  }

  v8f acc[4][2] = {};
  float4 bReg[4];

  // ---- prologue ----
#pragma unroll
  for (int rr = 0; rr < 2; ++rr)
    async_copy_b128(lds_off_of(&lA[0][(rr * 64 + arow) * LDS_STRIDE + achk]), aSrc[rr]);
#pragma unroll
  for (int rr = 0; rr < 4; ++rr)
    bReg[rr] = *(const float4*)(W + (size_t)(ncol0 + rr * 32 + rowq) * INTER + colq);
#pragma unroll
  for (int rr = 0; rr < 4; ++rr)
    cvt_store4(&lB[0][(rr * 32 + rowq) * LDS_STRIDE + colq], bReg[rr]);
  wait_async0();
  __syncthreads();

  int p = 0;
  for (int k0 = 0; k0 < INTER; k0 += KT) {
    const int kn = k0 + KT;
    const bool more = (kn < INTER);
    if (more) {
#pragma unroll
      for (int rr = 0; rr < 2; ++rr)
        async_copy_b128(lds_off_of(&lA[p ^ 1][(rr * 64 + arow) * LDS_STRIDE + achk]),
                        aSrc[rr] + kn);
#pragma unroll
      for (int rr = 0; rr < 4; ++rr) {
        const float* wp = W + (size_t)(ncol0 + rr * 32 + rowq) * INTER + kn + colq;
        bReg[rr] = *(const float4*)wp;
        __builtin_prefetch(wp + KT, 0, 0);
      }
    }
    v16bf af[4], bfr[2];
#pragma unroll
    for (int mf = 0; mf < 4; ++mf) af[mf] = load_a_frag(lA[p], wm * 64 + mf * 16, lane);
#pragma unroll
    for (int nf = 0; nf < 2; ++nf) bfr[nf] = load_b_frag(lB[p], wn * 32 + nf * 16, lane);
#pragma unroll
    for (int mf = 0; mf < 4; ++mf)
#pragma unroll
      for (int nf = 0; nf < 2; ++nf)
        acc[mf][nf] = __builtin_amdgcn_wmma_f32_16x16x32_bf16(
            false, af[mf], false, bfr[nf], (short)0, acc[mf][nf], false, false);
    if (more) {
#pragma unroll
      for (int rr = 0; rr < 4; ++rr)
        cvt_store4(&lB[p ^ 1][(rr * 32 + rowq) * LDS_STRIDE + colq], bReg[rr]);
    }
    wait_async0();
    __syncthreads();
    p ^= 1;
  }

  // scale by routing weight and scatter-add to out (token in <=2 buckets -> atomic)
#pragma unroll
  for (int mf = 0; mf < 4; ++mf) {
#pragma unroll
    for (int nf = 0; nf < 2; ++nf) {
      int n = ncol0 + wn * 32 + nf * 16 + (lane & 15);
#pragma unroll
      for (int r = 0; r < 8; ++r) {
        int m = wm * 64 + mf * 16 + r + ((lane >> 4) << 3);
        int gr = mbase + m;
        if (gr < rows) {
          int   t = rowToken[e * TOKENS + gr];
          float w = rowW[e * TOKENS + gr];
          atomicAdd(&out[(size_t)t * HIDDEN + n], w * acc[mf][nf][r]);
        }
      }
    }
  }
}

// ---------------- launcher ----------------
extern "C" void kernel_launch(void* const* d_in, const int* in_sizes, int n_in,
                              void* d_out, int out_size, void* d_ws, size_t ws_size,
                              hipStream_t stream) {
  const float* X     = (const float*)d_in[0];  // [TOKENS, HIDDEN]
  const int*   idx   = (const int*)  d_in[1];  // [TOKENS, 2]
  const float* wts   = (const float*)d_in[2];  // [TOKENS, 2]
  const float* Wup   = (const float*)d_in[3];  // [E, INTER, HIDDEN]
  const float* Wdown = (const float*)d_in[4];  // [E, HIDDEN, INTER]
  float* out = (float*)d_out;

  // workspace layout
  char* ws = (char*)d_ws;
  int*   cnt      = (int*)ws;                 ws += 256;
  int*   rowStart = (int*)ws;                 ws += 256;
  int*   rowToken = (int*)ws;                 ws += (size_t)NEXP * TOKENS * sizeof(int);
  float* rowW     = (float*)ws;               ws += (size_t)NEXP * TOKENS * sizeof(float);
  __bf16* Xb      = (__bf16*)ws;              ws += (size_t)TOKENS * HIDDEN * sizeof(__bf16);
  __bf16* Hbuf    = (__bf16*)ws;              // up to 16384 x INTER bf16 = 128 MB

  int nOut = TOKENS * HIDDEN;
  moe_zero_kernel<<<(nOut + 255) / 256, 256, 0, stream>>>(out, nOut, cnt);
  int n4 = (TOKENS * HIDDEN) / 4;
  moe_cvtx_kernel<<<(n4 + 255) / 256, 256, 0, stream>>>(X, Xb, n4);
  moe_route_kernel<<<(TOKENS + 255) / 256, 256, 0, stream>>>(idx, wts, cnt, rowToken, rowW);
  moe_prefix_kernel<<<1, 32, 0, stream>>>(cnt, rowStart);

  dim3 g1(NEXP, INTER / NT, TOKENS / MT);
  moe_up_kernel<<<g1, 256, 0, stream>>>(Xb, Wup, cnt, rowStart, rowToken, Hbuf);

  dim3 g2(NEXP, HIDDEN / NT, TOKENS / MT);
  moe_down_kernel<<<g2, 256, 0, stream>>>(Hbuf, Wdown, cnt, rowStart, rowToken, rowW, out);
}
